// _BaseSpiderConv_7851200217593
// MI455X (gfx1250) — matrix operand order repaired and checked by reading, not compiled, not run on hardware
//
#include <hip/hip_runtime.h>

typedef __attribute__((ext_vector_type(16))) _Float16 v16h;
typedef __attribute__((ext_vector_type(8)))  float    v8f;
typedef __attribute__((ext_vector_type(4)))  float    v4f;

constexpr int Bn = 16;     // batches
constexpr int Cn = 64;     // channels
constexpr int Nn = 2048;   // points
constexpr int Kn = 20;     // neighbors
constexpr int On = 64;     // out channels
constexpr int CT = 192;    // C*T
constexpr int CH = 120;    // reduction chunks = Kn * (CT/32)
constexpr size_t W2FRAG_BYTES = (size_t)CH * 4 * 32 * 16 * 2;  // 491,520 B

// ---------------------------------------------------------------------------
// Kernel 1: feat (B,C,N) -> featT (B,N,C), LDS-tiled 32x32 transpose so the
// main kernel's gather reads 64 contiguous floats per neighbor.
// ---------------------------------------------------------------------------
__global__ __launch_bounds__(256) void feat_transpose(const float* __restrict__ feat,
                                                      float* __restrict__ featT) {
  __shared__ float tile[32][33];
  const int b  = blockIdx.z;
  const int n0 = blockIdx.x * 32;
  const int c0 = blockIdx.y * 32;
  const int tx = threadIdx.x, ty = threadIdx.y;   // 32 x 8
#pragma unroll
  for (int i = 0; i < 32; i += 8)
    tile[ty + i][tx] = feat[((size_t)(b * Cn + c0 + ty + i)) * Nn + n0 + tx];
  __syncthreads();
#pragma unroll
  for (int i = 0; i < 32; i += 8)
    featT[((size_t)(b * Nn + n0 + ty + i)) * Cn + c0 + tx] = tile[tx][ty + i];
}

// ---------------------------------------------------------------------------
// Kernel 2: W2 (O, C*T, K) f32 -> per-lane WMMA-B fragments, f16.
// Reduction index r = k*192 + i, split into 120 chunks of 32.
// B-operand layout (v_wmma_*_16x16x32, wave32): lanes 0-15 hold K=0..7,16..23 of
// column n=lane; lanes 16-31 hold K=8..15,24..31 of column n=lane-16.
// Fragment buffer: [chunk][nt][lane][16 halves] -> each lane's v16h is one
// contiguous, 32B-aligned 32-byte read in the main loop.
// ---------------------------------------------------------------------------
__global__ __launch_bounds__(256) void w2_swizzle(const float* __restrict__ W2,
                                                  _Float16* __restrict__ w2frag) {
  const int e = blockIdx.x * 256 + threadIdx.x;   // 0 .. 245759
  const int h     = e & 15;
  const int L     = (e >> 4) & 31;
  const int nt    = (e >> 9) & 3;
  const int chunk = e >> 11;                      // 0 .. 119
  const int k  = chunk / 6;
  const int cc = chunk % 6;
  const int kk = (L < 16) ? (h < 8 ? h : 8 + h) : (h < 8 ? 8 + h : 16 + h);  // K within chunk
  const int o  = nt * 16 + (L & 15);
  const int i  = cc * 32 + kk;                    // 0 .. 191
  w2frag[e] = (_Float16)W2[((size_t)o * CT + i) * Kn + k];
}

// ---------------------------------------------------------------------------
// Kernel 3: fused gather + Taylor + GEMM (WMMA) + bias + BN + ReLU.
// Grid: 256 blocks x 256 threads. Block = (batch, 128-point tile); each of the
// 8 waves owns 16 points and accumulates all 64 outputs in 4 WMMA C tiles.
// Per chunk: load 4 B fragments first, then build A (VALU overlaps the load
// latency), then 4 back-to-back WMMAs; idx for k+1 is prefetched to break the
// idx -> gather serial chain.
// ---------------------------------------------------------------------------
__global__ __launch_bounds__(256) void spider_wmma(
    const float* __restrict__ featT, const _Float16* __restrict__ w2frag,
    const int* __restrict__ idx, const float* __restrict__ gp,
    const float* __restrict__ W1, const float* __restrict__ b1v,
    const float* __restrict__ b2v, const float* __restrict__ gam,
    const float* __restrict__ bet, const float* __restrict__ rmean,
    const float* __restrict__ rvar, float* __restrict__ out) {
  const int lane = threadIdx.x & 31;
  const int wave = threadIdx.x >> 5;
  const int b    = blockIdx.x >> 4;
  const int tl   = blockIdx.x & 15;
  const int n0   = tl * 128 + wave * 16;
  const int m    = lane & 15;          // this lane's point row within the tile
  const int n    = n0 + m;
  const bool lo  = lane < 16;

  v8f acc0 = {}, acc1 = {}, acc2 = {}, acc3 = {};
  const float* gpb  = gp + (size_t)b * 3 * Nn * Kn;
  const int*   idxp = idx + ((size_t)b * Nn + n) * Kn;

  int gi = idxp[0];                    // neighbor index for k=0 (prefetched)

  for (int k = 0; k < Kn; ++k) {
    // ---- gather: 64 channels of this point's k-th neighbor (L2-resident) ----
    const v4f* fp = (const v4f*)(featT + ((size_t)b * Nn + gi) * Cn);
    float gfr[64];
#pragma unroll
    for (int q = 0; q < 16; ++q) {
      v4f t = fp[q];
      gfr[4 * q + 0] = t[0]; gfr[4 * q + 1] = t[1];
      gfr[4 * q + 2] = t[2]; gfr[4 * q + 3] = t[3];
    }
    // prefetch next k's neighbor index (clamped; branch-free)
    const int kn = (k + 1 < Kn) ? (k + 1) : (Kn - 1);
    const int gi_next = idxp[kn];

    // ---- Taylor coefficients (W1/b1 are uniform -> SGPR loads) ----
    const float X = gpb[(size_t)(0 * Nn + n) * Kn + k];
    const float Y = gpb[(size_t)(1 * Nn + n) * Kn + k];
    const float Z = gpb[(size_t)(2 * Nn + n) * Kn + k];
    const float XX = X * X, YY = Y * Y, ZZ = Z * Z;
    const float XY = X * Y, XZ = X * Z, YZ = Y * Z;
    const float tm[19] = {X, Y, Z, XX, YY, ZZ, XX * X, YY * Y, ZZ * Z,
                          XY, XZ, YZ, X * XY, X * XZ, Y * YZ, Y * XY,
                          Z * XZ, Z * YZ, XY * Z};
    float tay[3];
#pragma unroll
    for (int t = 0; t < 3; ++t) {
      float s = b1v[t];
#pragma unroll
      for (int j = 0; j < 19; ++j) s += W1[t * 19 + j] * tm[j];
      tay[t] = s;
    }

    // ---- 6 reduction chunks of 32 (i = c*3+t), 4 WMMAs each ----
#pragma unroll
    for (int c6 = 0; c6 < 6; ++c6) {
      // 1) issue all four B-fragment loads for this chunk up front
      const int chunk = k * 6 + c6;
      const v16h* bp = ((const v16h*)w2frag) + (size_t)(chunk * 4) * 32 + lane;
      const v16h bf0 = bp[0];
      const v16h bf1 = bp[32];
      const v16h bf2 = bp[64];
      const v16h bf3 = bp[96];

      // 2) build the A fragment in ISA register layout (pure VALU, overlaps
      //    the loads above):
      // lanes 0-15 : halves 0-7 -> K 0-7,  halves 8-15 -> K 16-23
      // lanes 16-31: halves 0-7 -> K 8-15, halves 8-15 -> K 24-31
      v16h a;
      if (lo) {
#pragma unroll
        for (int j = 0; j < 8; ++j) {
          const int i0 = c6 * 32 + j;
          const int i1 = c6 * 32 + 16 + j;
          a[j]     = (_Float16)(gfr[i0 / 3] * tay[i0 % 3]);
          a[8 + j] = (_Float16)(gfr[i1 / 3] * tay[i1 % 3]);
        }
      } else {
#pragma unroll
        for (int j = 0; j < 8; ++j) {
          const int i0 = c6 * 32 + 8 + j;
          const int i1 = c6 * 32 + 24 + j;
          a[j]     = (_Float16)(gfr[i0 / 3] * tay[i0 % 3]);
          a[8 + j] = (_Float16)(gfr[i1 / 3] * tay[i1 % 3]);
        }
      }

      // 3) four back-to-back WMMAs (EXEC all-1s here; accumulate in f32)
      acc0 = __builtin_amdgcn_wmma_f32_16x16x32_f16(false, a, false, bf0,
                                                    (short)0, acc0, false, false);
      acc1 = __builtin_amdgcn_wmma_f32_16x16x32_f16(false, a, false, bf1,
                                                    (short)0, acc1, false, false);
      acc2 = __builtin_amdgcn_wmma_f32_16x16x32_f16(false, a, false, bf2,
                                                    (short)0, acc2, false, false);
      acc3 = __builtin_amdgcn_wmma_f32_16x16x32_f16(false, a, false, bf3,
                                                    (short)0, acc3, false, false);
    }
    gi = gi_next;
  }

  // ---- epilogue: bias + inference BN + ReLU ----
  // C/D layout: VGPR v -> row M = v + 8*(lane>=16), lane%16 -> column o.
  const int rowoff = (lane >> 4) * 8;
#pragma unroll
  for (int nt = 0; nt < 4; ++nt) {
    const v8f acc = (nt == 0) ? acc0 : (nt == 1) ? acc1 : (nt == 2) ? acc2 : acc3;
    const int o = nt * 16 + m;
    const float bias = b2v[o];
    const float inv  = gam[o] * rsqrtf(rvar[o] + 1e-5f);
    const float mu   = rmean[o];
    const float bb   = bet[o];
    float* op = out + ((size_t)b * On + o) * Nn + n0 + rowoff;
#pragma unroll
    for (int v = 0; v < 8; ++v) {
      const float val = (acc[v] + bias - mu) * inv + bb;
      op[v] = val > 0.f ? val : 0.f;
    }
  }
}

// ---------------------------------------------------------------------------
extern "C" void kernel_launch(void* const* d_in, const int* in_sizes, int n_in,
                              void* d_out, int out_size, void* d_ws, size_t ws_size,
                              hipStream_t stream) {
  const float* feat  = (const float*)d_in[0];
  const int*   idx   = (const int*)d_in[1];
  const float* gp    = (const float*)d_in[2];
  const float* W1    = (const float*)d_in[3];
  const float* b1v   = (const float*)d_in[4];
  const float* W2    = (const float*)d_in[5];
  const float* b2v   = (const float*)d_in[6];
  const float* gam   = (const float*)d_in[7];
  const float* bet   = (const float*)d_in[8];
  const float* rmean = (const float*)d_in[9];
  const float* rvar  = (const float*)d_in[10];
  float* out = (float*)d_out;

  // workspace: [0, 491520) = W2 fragments (f16); then featT (B*N*C f32, 8 MB)
  _Float16* w2frag = (_Float16*)d_ws;
  float* featT = (float*)((char*)d_ws + W2FRAG_BYTES);

  feat_transpose<<<dim3(Nn / 32, Cn / 32, Bn), dim3(32, 8), 0, stream>>>(feat, featT);
  w2_swizzle<<<(CH * 4 * 32 * 16) / 256, 256, 0, stream>>>(W2, w2frag);
  spider_wmma<<<Bn * (Nn / 128), 256, 0, stream>>>(featT, w2frag, idx, gp, W1, b1v,
                                                   b2v, gam, bet, rmean, rvar, out);
}
